// OSRENet_39840116638142
// MI455X (gfx1250) — matrix operand order repaired and checked by reading, not compiled, not run on hardware
//
#include <hip/hip_runtime.h>

// ---------------- CDNA5 vector types ----------------
typedef __attribute__((ext_vector_type(16))) __bf16          v16bf;
typedef __attribute__((ext_vector_type(8)))  float           v8f;
typedef __attribute__((ext_vector_type(4)))  float           v4f;
typedef __attribute__((ext_vector_type(4)))  unsigned short  u16x4;
typedef __attribute__((ext_vector_type(8)))  unsigned short  u16x8;
typedef __attribute__((ext_vector_type(16))) unsigned short  u16x16;

// ---------------- problem constants ----------------
#define HID   64
#define SEQ   60
#define FEAT  40
#define GDIM  256          // 4*HID
#define KCAT  128          // padded [x|h] concat width
#define BT    16           // batch rows per workgroup
#define AST   136          // LDS A-row stride in halfs (272B -> conflict free)
#define GST   264          // LDS gate-row stride in floats

__device__ __forceinline__ unsigned short f2bf(float f) {
    unsigned u = __builtin_bit_cast(unsigned, f);
    return (unsigned short)((u + 0x7FFFu + ((u >> 16) & 1u)) >> 16);
}
// branch-free fast activations: v_exp_f32 + v_rcp_f32, no libm divergence
__device__ __forceinline__ float sigm(float v) {
    return __builtin_amdgcn_rcpf(1.0f + __expf(-v));
}
__device__ __forceinline__ float tanh_fast(float v) {
    float a = __builtin_fabsf(v);
    float t = __expf(-2.0f * a);
    float r = (1.0f - t) * __builtin_amdgcn_rcpf(1.0f + t);
    return __builtin_copysignf(r, v);
}

// ---------------- weight / bias prep: f32 -> bf16 concatenated [256 x 128] ----------------
__global__ void lstm_prep_kernel(const float* __restrict__ Wih0, const float* __restrict__ Whh0,
                                 const float* __restrict__ bih0, const float* __restrict__ bhh0,
                                 const float* __restrict__ Wih1, const float* __restrict__ Whh1,
                                 const float* __restrict__ bih1, const float* __restrict__ bhh1,
                                 unsigned short* __restrict__ w0, unsigned short* __restrict__ w1,
                                 float* __restrict__ bias0, float* __restrict__ bias1) {
    int idx = blockIdx.x * 256 + threadIdx.x;      // 0 .. 32767
    if (idx < GDIM * KCAT) {
        int g = idx >> 7, k = idx & 127;
        float v0 = (k < FEAT) ? Wih0[g * FEAT + k] : ((k < HID) ? 0.0f : Whh0[g * HID + (k - HID)]);
        float v1 = (k < HID) ? Wih1[g * HID + k] : Whh1[g * HID + (k - HID)];
        w0[idx] = f2bf(v0);
        w1[idx] = f2bf(v1);
        if (k == 0) {
            bias0[g] = bih0[g] + bhh0[g];
            bias1[g] = bih1[g] + bhh1[g];
        }
    }
}

// one fused GEMM step: gates[16 x 256] = A[16 x 128] * Bcat + bias, B held in VGPRs
__device__ __forceinline__ void gemm_step(const unsigned short* As,
                                          const v16bf (&Bfr)[2][4],
                                          float bv0, float bv1,
                                          float* gs, int wave, int lane) {
    const int hi = lane >> 4, ln = lane & 15;
    // A fragments: row = ln, chunks at K = 32*kc + 16*hi and +8
    v16bf A[4];
#pragma unroll
    for (int kc = 0; kc < 4; ++kc) {
        const unsigned short* p = As + ln * AST + kc * 32 + 16 * hi;
        u16x8 lo  = *(const u16x8*)(p);
        u16x8 hi8 = *(const u16x8*)(p + 8);
        u16x16 cat = __builtin_shufflevector(lo, hi8, 0, 1, 2, 3, 4, 5, 6, 7,
                                                       8, 9, 10, 11, 12, 13, 14, 15);
        A[kc] = __builtin_bit_cast(v16bf, cat);
    }
#pragma unroll
    for (int tl = 0; tl < 2; ++tl) {
        float bv = tl ? bv1 : bv0;
        v8f acc;
#pragma unroll
        for (int i = 0; i < 8; ++i) acc[i] = bv;
#pragma unroll
        for (int kc = 0; kc < 4; ++kc) {
            acc = __builtin_amdgcn_wmma_f32_16x16x32_bf16(
                false, A[kc], false, Bfr[tl][kc], (short)0, acc, false, false);
        }
        const int col = (2 * wave + tl) * 16 + ln;
#pragma unroll
        for (int i = 0; i < 8; ++i) gs[(i + 8 * hi) * GST + col] = acc[i];
    }
}

// ---------------- fused 2-layer LSTM + FC, 16 batch rows per workgroup ----------------
__global__ void __launch_bounds__(256)
lstm_fused_kernel(const float* __restrict__ x,
                  const unsigned short* __restrict__ w0, const unsigned short* __restrict__ w1,
                  const float* __restrict__ bias0, const float* __restrict__ bias1,
                  const float* __restrict__ fcw, const float* __restrict__ fcb,
                  float* __restrict__ out) {
    __shared__ __align__(16) unsigned short A0s[BT * AST];  // [x_t | h0_{t-1}] bf16
    __shared__ __align__(16) unsigned short A1s[BT * AST];  // [h0_t | h1_{t-1}] bf16
    __shared__ __align__(16) float gs[BT * GST];            // gate staging f32

    const int tid  = threadIdx.x;
    const int lane = tid & 31;
    const int wave = tid >> 5;
    const int hi   = lane >> 4;
    const int ln   = lane & 15;
    const int b0   = blockIdx.x * BT;

    // ---- preload loop-invariant B fragments (both layers) into VGPRs ----
    v16bf Bf[2][2][4];
    float bv[2][2];
    {
        const unsigned short* wsrc[2] = { w0, w1 };
        const float* bsrc[2] = { bias0, bias1 };
#pragma unroll
        for (int L = 0; L < 2; ++L) {
#pragma unroll
            for (int tl = 0; tl < 2; ++tl) {
                const int g = (2 * wave + tl) * 16 + ln;   // gate column for this lane
                bv[L][tl] = bsrc[L][g];
#pragma unroll
                for (int kc = 0; kc < 4; ++kc) {
                    Bf[L][tl][kc] = *(const v16bf*)(wsrc[L] + g * KCAT + kc * 32 + hi * 16);
                }
            }
        }
    }

    // ---- zero-init both A buffers (also guarantees zero pad cols 40..63: NaN*0=NaN in WMMA) ----
    for (int idx = tid; idx < BT * AST / 2; idx += 256) {
        ((unsigned*)A0s)[idx] = 0u;
        ((unsigned*)A1s)[idx] = 0u;
    }

    // ---- per-thread x staging setup: 160 threads move one float4 each per step ----
    const bool xload = tid < (BT * FEAT / 4);          // 160 threads
    const int  xr = tid / 10;                          // batch row within tile
    const int  xq = tid - xr * 10;                     // float4 index within row
    const float* xptr = x + (size_t)(b0 + xr) * (SEQ * FEAT) + xq * 4;
    unsigned short* a0x = A0s + xr * AST + xq * 4;

    const int jj = tid & 63;   // hidden unit handled by this thread
    const int r0 = tid >> 6;   // base batch row
    float c0[4] = {0.f, 0.f, 0.f, 0.f};
    float c1[4] = {0.f, 0.f, 0.f, 0.f};
    float facc[4] = {0.f, 0.f, 0.f, 0.f};

#pragma unroll 1
    for (int t = 0; t < SEQ; ++t) {
        // (a) stage x_t into A0 (vectorized, branch-light) + prefetch x_{t+1}
        if (xload) {
            v4f v = *(const v4f*)xptr;
            u16x4 h4;
            h4[0] = f2bf(v[0]); h4[1] = f2bf(v[1]);
            h4[2] = f2bf(v[2]); h4[3] = f2bf(v[3]);
            *(u16x4*)a0x = h4;
            __builtin_prefetch(xptr + FEAT, 0, 3);     // global_prefetch_b8 for t+1
            xptr += FEAT;
        }
        __syncthreads();

        // (b) layer-0 fused GEMM
        gemm_step(A0s, Bf[0], bv[0][0], bv[0][1], gs, wave, lane);
        __syncthreads();

        // (c) layer-0 gates -> h0, c0 ; feed A1 (this t) and A0 (next t)
#pragma unroll
        for (int q = 0; q < 4; ++q) {
            int r = 4 * q + r0;
            float gi = gs[r * GST + jj];
            float gf = gs[r * GST + HID + jj];
            float gg = gs[r * GST + 2 * HID + jj];
            float go = gs[r * GST + 3 * HID + jj];
            float c  = sigm(gf) * c0[q] + sigm(gi) * tanh_fast(gg);
            c0[q] = c;
            unsigned short hb = f2bf(sigm(go) * tanh_fast(c));
            A1s[r * AST + jj] = hb;
            A0s[r * AST + HID + jj] = hb;
        }
        __syncthreads();

        // (d) layer-1 fused GEMM
        gemm_step(A1s, Bf[1], bv[1][0], bv[1][1], gs, wave, lane);
        __syncthreads();

        // (e) layer-1 gates -> h1, c1 ; fold FC accumulation in-place
        float wfc = fcw[t * HID + jj];
#pragma unroll
        for (int q = 0; q < 4; ++q) {
            int r = 4 * q + r0;
            float gi = gs[r * GST + jj];
            float gf = gs[r * GST + HID + jj];
            float gg = gs[r * GST + 2 * HID + jj];
            float go = gs[r * GST + 3 * HID + jj];
            float c  = sigm(gf) * c1[q] + sigm(gi) * tanh_fast(gg);
            c1[q] = c;
            float h = sigm(go) * tanh_fast(c);
            A1s[r * AST + HID + jj] = f2bf(h);
            facc[q] += h * wfc;
        }
        // no barrier needed: next iter's (a) touches a disjoint LDS region,
        // and every A1/gs consumer re-synchronizes before reuse
    }

    // ---- FC reduction across hidden units ----
    __syncthreads();   // protect gs vs last EW reads
#pragma unroll
    for (int q = 0; q < 4; ++q) gs[(4 * q + r0) * 66 + jj] = facc[q];
    __syncthreads();
    if (tid < BT) {
        float s = 0.f;
#pragma unroll
        for (int j = 0; j < HID; ++j) s += gs[tid * 66 + j];
        out[b0 + tid] = s + fcb[0];
    }
}

extern "C" void kernel_launch(void* const* d_in, const int* in_sizes, int n_in,
                              void* d_out, int out_size, void* d_ws, size_t ws_size,
                              hipStream_t stream) {
    (void)in_sizes; (void)n_in; (void)out_size; (void)ws_size;
    const float* x    = (const float*)d_in[0];
    const float* Wih0 = (const float*)d_in[1];
    const float* Whh0 = (const float*)d_in[2];
    const float* bih0 = (const float*)d_in[3];
    const float* bhh0 = (const float*)d_in[4];
    const float* Wih1 = (const float*)d_in[5];
    const float* Whh1 = (const float*)d_in[6];
    const float* bih1 = (const float*)d_in[7];
    const float* bhh1 = (const float*)d_in[8];
    const float* fcw  = (const float*)d_in[9];
    const float* fcb  = (const float*)d_in[10];

    unsigned short* w0 = (unsigned short*)d_ws;                       // 256*128 bf16
    unsigned short* w1 = w0 + GDIM * KCAT;                            // 256*128 bf16
    float* bias0 = (float*)((char*)d_ws + 2 * GDIM * KCAT * sizeof(unsigned short));
    float* bias1 = bias0 + GDIM;

    lstm_prep_kernel<<<128, 256, 0, stream>>>(Wih0, Whh0, bih0, bhh0,
                                              Wih1, Whh1, bih1, bhh1,
                                              w0, w1, bias0, bias1);
    lstm_fused_kernel<<<8192 / BT, 256, 0, stream>>>(x, w0, w1, bias0, bias1,
                                                     fcw, fcb, (float*)d_out);
}
